// ConstantQResonantPacket_31653908972266
// MI455X (gfx1250) — matrix operand order repaired and checked by reading, not compiled, not run on hardware
//
#include <hip/hip_runtime.h>
#include <hip/hip_bf16.h>

typedef _Float16 half8  __attribute__((ext_vector_type(8)));
typedef _Float16 half16 __attribute__((ext_vector_type(16)));
typedef float    v8f    __attribute__((ext_vector_type(8)));
typedef float    f2     __attribute__((ext_vector_type(2)));

constexpr int BTOT = 32768;   // batch rows
constexpr int DDIM = 512;     // feature dim
constexpr int KDIM = 1024;    // packets
constexpr int BM = 64;        // block tile rows
constexpr int BN = 128;       // block tile cols
constexpr int KC = 32;        // k-step (WMMA depth)
constexpr int LDT = 40;       // LDS tile stride in halves (80B, 16B aligned, padded)
constexpr int SZ_X = BM * LDT;               // 2560 halves
constexpr int SZ_W = BN * LDT;               // 5120 halves
constexpr int BUFH = 2 * SZ_X + 3 * SZ_W;    // 20480 halves = 40 KB per stage
constexpr int NSTEP = DDIM / KC;             // 16 k-steps

#if defined(__AMDGCN__) && __has_builtin(__builtin_amdgcn_global_load_async_to_lds_b128) && \
    __has_builtin(__builtin_amdgcn_s_wait_asynccnt)
#define CQRP_ASYNC 1
#else
#define CQRP_ASYNC 0
#endif

#if CQRP_ASYNC
// builtin signature (from clang diagnostic): (int4 __global*, int4 __shared*, Imm int, Imm int)
typedef int v4i_gcc __attribute__((vector_size(16)));
typedef __attribute__((address_space(1))) v4i_gcc as1_v4i;
typedef __attribute__((address_space(3))) v4i_gcc as3_v4i;
#endif

// stage one 16-byte chunk: global -> LDS
static __device__ inline void cp16(_Float16* lds, const _Float16* g) {
#if CQRP_ASYNC
    __builtin_amdgcn_global_load_async_to_lds_b128((as1_v4i*)g, (as3_v4i*)lds,
                                                   /*offset=*/0, /*cpol=*/0);
#else
    *(half8*)lds = *(const half8*)g;
#endif
}

// ---------------- prep kernels ----------------

__global__ void cqrp_prep_x(const float* __restrict__ x,
                            _Float16* __restrict__ xh, _Float16* __restrict__ xl,
                            float* __restrict__ xsq) {
    __shared__ float red[256];
    const int b = blockIdx.x;
    const int t = threadIdx.x;
    float s = 0.f;
    for (int d = t; d < DDIM; d += 256) {
        float v = __builtin_nontemporal_load(&x[b * DDIM + d]);
        _Float16 hi = (_Float16)v;
        _Float16 lo = (_Float16)(v - (float)hi);
        xh[b * DDIM + d] = hi;
        xl[b * DDIM + d] = lo;
        s += v * v;
    }
    red[t] = s;
    __syncthreads();
    for (int o = 128; o > 0; o >>= 1) {
        if (t < o) red[t] += red[t + o];
        __syncthreads();
    }
    if (t == 0) xsq[b] = red[0];
}

__global__ void cqrp_prep_k(const float* __restrict__ omega, const float* __restrict__ centers,
                            _Float16* __restrict__ wh, _Float16* __restrict__ wl,
                            _Float16* __restrict__ ch,
                            float* __restrict__ csq, float* __restrict__ inv2s2) {
    __shared__ float rw[256];
    __shared__ float rc[256];
    const int k = blockIdx.x;
    const int t = threadIdx.x;
    float sw = 0.f, sc = 0.f;
    for (int d = t; d < DDIM; d += 256) {
        float w = __builtin_nontemporal_load(&omega[k * DDIM + d]);
        float c = __builtin_nontemporal_load(&centers[k * DDIM + d]);
        _Float16 hi = (_Float16)w;
        wh[k * DDIM + d] = hi;
        wl[k * DDIM + d] = (_Float16)(w - (float)hi);
        ch[k * DDIM + d] = (_Float16)c;
        sw += w * w;
        sc += c * c;
    }
    rw[t] = sw; rc[t] = sc;
    __syncthreads();
    for (int o = 128; o > 0; o >>= 1) {
        if (t < o) { rw[t] += rw[t + o]; rc[t] += rc[t + o]; }
        __syncthreads();
    }
    if (t == 0) {
        csq[k] = rc[0];
        float sigma = rw[0] + 1e-4f;          // Q_FACTOR=1: sigma = ||omega||^2 + eps
        inv2s2[k] = 0.5f / (sigma * sigma);   // envelope = exp(-d2 * inv2s2)
    }
}

// ---------------- fused WMMA kernel ----------------

static __device__ inline half16 ld_frag2(const _Float16* p0, const _Float16* p1) {
    half8 a = *(const half8*)p0;
    half8 b = *(const half8*)p1;
    return __builtin_shufflevector(a, b, 0, 1, 2, 3, 4, 5, 6, 7,
                                         8, 9, 10, 11, 12, 13, 14, 15);
}

__global__ __launch_bounds__(256) void cqrp_fused(
    const _Float16* __restrict__ xh, const _Float16* __restrict__ xl,
    const _Float16* __restrict__ wh, const _Float16* __restrict__ wl,
    const _Float16* __restrict__ ch,
    const float* __restrict__ xsq, const float* __restrict__ csq,
    const float* __restrict__ inv2s2, const float* __restrict__ phase,
    const float* __restrict__ ar, const float* __restrict__ ai,
    f2* __restrict__ out) {
    extern __shared__ _Float16 smem[];   // 2 stages x 40 KB

    const int t    = threadIdx.x;
    const int wave = t >> 5;
    const int lane = t & 31;
    const int h    = lane >> 4;          // which half of the wave
    const int ln   = lane & 15;
    const int rt     = wave >> 1;        // row tile 0..3
    const int ctBase = (wave & 1) * 4;   // col tile base: 0 or 4
    const int rowBase = blockIdx.y * BM;
    const int colBase = blockIdx.x * BN;

    v8f zero = {};
    v8f accP[4], accD[4];
#pragma unroll
    for (int i = 0; i < 4; ++i) { accP[i] = zero; accD[i] = zero; }

    const int xRow = t >> 2, xChk = t & 3;   // X tile: 64 rows x 4 chunks of 8 halves

    // issue all async copies for one stage: 8 per thread (2 for X hi/lo, 6 for W/C)
    auto stage = [&](int buf, int kk) {
        _Float16* bXh = smem + buf * BUFH;
        _Float16* bXl = bXh + SZ_X;
        _Float16* bWh = bXl + SZ_X;
        _Float16* bWl = bWh + SZ_W;
        _Float16* bCh = bWl + SZ_W;
        cp16(&bXh[xRow * LDT + xChk * 8],
             &xh[(size_t)(rowBase + xRow) * DDIM + kk + xChk * 8]);
        cp16(&bXl[xRow * LDT + xChk * 8],
             &xl[(size_t)(rowBase + xRow) * DDIM + kk + xChk * 8]);
#pragma unroll
        for (int i = 0; i < 2; ++i) {
            int idx = t + i * 256;
            int r = idx >> 2, cc = idx & 3;
            cp16(&bWh[r * LDT + cc * 8],
                 &wh[(size_t)(colBase + r) * DDIM + kk + cc * 8]);
            cp16(&bWl[r * LDT + cc * 8],
                 &wl[(size_t)(colBase + r) * DDIM + kk + cc * 8]);
            cp16(&bCh[r * LDT + cc * 8],
                 &ch[(size_t)(colBase + r) * DDIM + kk + cc * 8]);
        }
    };

    stage(0, 0);   // prologue

#pragma unroll 1
    for (int i = 0; i < NSTEP; ++i) {
        if (i + 1 < NSTEP) {
            stage((i + 1) & 1, (i + 1) * KC);   // overlap next stage with this compute
#if CQRP_ASYNC
            __builtin_amdgcn_s_wait_asynccnt(8);  // current stage's 8 done (in-order)
#endif
        } else {
#if CQRP_ASYNC
            __builtin_amdgcn_s_wait_asynccnt(0);
#endif
        }
        __syncthreads();   // make this stage visible to all waves

        const _Float16* base = smem + (i & 1) * BUFH;
        const _Float16* bXh = base;
        const _Float16* bXl = bXh + SZ_X;
        const _Float16* bWh = bXl + SZ_X;
        const _Float16* bWl = bWh + SZ_W;
        const _Float16* bCh = bWl + SZ_W;

        // A fragments (16x32 f16: VGPR0-3 hold K=8h..8h+7, VGPR4-7 hold K=16+8h..+7)
        const int aOff = (rt * 16 + ln) * LDT;
        half16 aH = ld_frag2(&bXh[aOff + 8 * h], &bXh[aOff + 16 + 8 * h]);
        half16 aL = ld_frag2(&bXl[aOff + 8 * h], &bXl[aOff + 16 + 8 * h]);

#pragma unroll
        for (int ct = 0; ct < 4; ++ct) {
            // B fragment (32x16): column n = ln holds K = 8h..8h+7 then 16+8h..16+8h+7
            const int bOff = ((ctBase + ct) * 16 + ln) * LDT + 8 * h;
            half16 fWh = ld_frag2(&bWh[bOff], &bWh[bOff + 16]);
            half16 fWl = ld_frag2(&bWl[bOff], &bWl[bOff + 16]);
            half16 fC  = ld_frag2(&bCh[bOff], &bCh[bOff + 16]);

            // phase GEMM with hi/lo split: xh*wh + xh*wl + xl*wh  (near-f32 accuracy)
            accP[ct] = __builtin_amdgcn_wmma_f32_16x16x32_f16(
                false, aH, false, fWh, (short)0, accP[ct], false, false);
            accP[ct] = __builtin_amdgcn_wmma_f32_16x16x32_f16(
                false, aH, false, fWl, (short)0, accP[ct], false, false);
            accP[ct] = __builtin_amdgcn_wmma_f32_16x16x32_f16(
                false, aL, false, fWh, (short)0, accP[ct], false, false);
            // distance GEMM: envelope = exp(-d2/2sigma^2), sigma^2 ~ 2e7 -> f16 is plenty
            accD[ct] = __builtin_amdgcn_wmma_f32_16x16x32_f16(
                false, aH, false, fC, (short)0, accD[ct], false, false);
        }
        __syncthreads();   // all waves done reading this buffer before it is restaged
    }

    // ---------------- epilogue ----------------
    float xs[8];
#pragma unroll
    for (int r = 0; r < 8; ++r)
        xs[r] = xsq[rowBase + rt * 16 + 8 * h + r];   // C/D layout: M = r + 8h

#pragma unroll
    for (int ct = 0; ct < 4; ++ct) {
        const int gk = colBase + (ctBase + ct) * 16 + ln;
        const float cs  = csq[gk];
        const float is2 = inv2s2[gk];
        const float pv  = phase[gk];
        const float arv = ar[gk];
        const float aiv = ai[gk];
#pragma unroll
        for (int r = 0; r < 8; ++r) {
            const int gb = rowBase + rt * 16 + 8 * h + r;
            float d2  = fmaxf(xs[r] + cs - 2.0f * accD[ct][r], 0.0f);
            float env = __expf(-d2 * is2);
            float s, c;
            __sincosf(accP[ct][r] + pv, &s, &c);
            f2 val;
            val.x = env * (c * arv - s * aiv);
            val.y = env * (c * aiv + s * arv);
            // streaming 268 MB output: non-temporal so x/omega tiles stay L2-resident
            __builtin_nontemporal_store(val, &out[(size_t)gb * KDIM + gk]);
        }
    }
}

// ---------------- host launcher ----------------

extern "C" void kernel_launch(void* const* d_in, const int* in_sizes, int n_in,
                              void* d_out, int out_size, void* d_ws, size_t ws_size,
                              hipStream_t stream) {
    (void)in_sizes; (void)n_in; (void)out_size; (void)ws_size;
    const float* x       = (const float*)d_in[0];
    const float* omega   = (const float*)d_in[1];
    const float* phase   = (const float*)d_in[2];
    const float* ampr    = (const float*)d_in[3];
    const float* ampi    = (const float*)d_in[4];
    const float* centers = (const float*)d_in[5];

    char* ws = (char*)d_ws;
    size_t off = 0;
    auto carve = [&](size_t bytes) -> void* {
        void* p = ws + off;
        off = (off + bytes + 255) & ~(size_t)255;
        return p;
    };
    _Float16* xh  = (_Float16*)carve((size_t)BTOT * DDIM * sizeof(_Float16));
    _Float16* xl  = (_Float16*)carve((size_t)BTOT * DDIM * sizeof(_Float16));
    _Float16* wh  = (_Float16*)carve((size_t)KDIM * DDIM * sizeof(_Float16));
    _Float16* wl  = (_Float16*)carve((size_t)KDIM * DDIM * sizeof(_Float16));
    _Float16* chv = (_Float16*)carve((size_t)KDIM * DDIM * sizeof(_Float16));
    float* xsq    = (float*)carve((size_t)BTOT * sizeof(float));
    float* csq    = (float*)carve((size_t)KDIM * sizeof(float));
    float* i2s2   = (float*)carve((size_t)KDIM * sizeof(float));

    cqrp_prep_x<<<BTOT, 256, 0, stream>>>(x, xh, xl, xsq);
    cqrp_prep_k<<<KDIM, 256, 0, stream>>>(omega, centers, wh, wl, chv, csq, i2s2);

    dim3 grid(KDIM / BN, BTOT / BM);
    size_t ldsBytes = (size_t)2 * BUFH * sizeof(_Float16);   // 81920 B
    cqrp_fused<<<grid, 256, ldsBytes, stream>>>(xh, xl, wh, wl, chv, xsq, csq, i2s2,
                                                phase, ampr, ampi, (f2*)d_out);
}
